// MISA_11725260718233
// MI455X (gfx1250) — matrix operand order repaired and checked by reading, not compiled, not run on hardware
//
#include <hip/hip_runtime.h>
#include <math.h>

#define B_SZ 4096
#define DIN  1024
#define DOUT 1024
#define NEXP 8

typedef __bf16 bf16_t;
typedef __bf16 v16bf __attribute__((ext_vector_type(16)));
typedef float  v8f   __attribute__((ext_vector_type(8)));

// ---------------- activation helpers ----------------
template<int ACT> __device__ __forceinline__ float activate(float v) {
    if constexpr (ACT == 1)      return fmaxf(v, 0.0f);                       // relu
    else if constexpr (ACT == 2) return tanhf(v);                             // tanh
    else if constexpr (ACT == 3) return v / (1.0f + expf(-v));                // silu
    else if constexpr (ACT == 4) return 0.5f * v * (1.0f + erff(v * 0.70710678118654752f)); // gelu exact
    else                         return v;
}

// ---------------- WMMA GEMM ----------------
// C = act( A(bf16)[M,K] @ W(f32)[K,N] + bias [+ extraX[row]*extraW[col]] )
// MODE 0: outF = val ; MODE 1: outB = val (bf16) ; MODE 2: outF += wvec[row]*val
// Double-buffered LDS; A tile moved with CDNA5 async global->LDS copies
// (ASYNCcnt), W tile prefetched to VGPRs and converted f32->bf16 transposed.
constexpr int BM = 128, BN = 128, BK = 32;
constexpr int LSTR = 40;   // padded K-stride in bf16 elems (80B, multiple of 16B)

template<int ACT, int MODE, int EXTRA>
__global__ __launch_bounds__(256, 1)
void gemm_bf16_wmma(const bf16_t* __restrict__ A, int lda,
                    const float* __restrict__ W, int ldw,
                    const float* __restrict__ bias,
                    float* __restrict__ outF, bf16_t* __restrict__ outB, int ldo,
                    const float* __restrict__ wvec, int wstride, int woff,
                    const float* __restrict__ extraW, const float* __restrict__ extraX,
                    int Ktot)
{
    __shared__ __align__(16) bf16_t lsA[2][BM * LSTR];
    __shared__ __align__(16) bf16_t lsB[2][BN * LSTR];

    const int bm0  = blockIdx.y * BM;
    const int bn0  = blockIdx.x * BN;
    const int tid  = threadIdx.x;
    const int wid  = tid >> 5;
    const int lane = tid & 31;
    const int wm   = wid & 3;      // 4 waves along M (32 rows each)
    const int wn   = wid >> 2;     // 2 waves along N (64 cols each)
    const int half = lane >> 4;    // lane 0-15 vs 16-31
    const int mr   = lane & 15;

    // A-tile async copy: thread -> (row, 32B segment)
    const int arow = tid >> 1, aseg = tid & 1;
    const unsigned aVoff = (unsigned)(arow * lda + aseg * 16) * 2u;  // bytes
    // W-tile: 4 float4 per thread
    int wk[4], wnl[4];
    for (int it = 0; it < 4; ++it) {
        const int q = tid + it * 256;
        wk[it]  = q >> 5;
        wnl[it] = (q & 31) << 2;
    }

    auto issueA = [&](int k0, int buf) {
        const bf16_t* sA = A + (size_t)bm0 * lda + k0;                 // uniform
        unsigned ldsOff = (unsigned)(size_t)(&lsA[buf][arow * LSTR + aseg * 16]);
        asm volatile("global_load_async_to_lds_b128 %0, %1, %2 offset:0\n\t"
                     "global_load_async_to_lds_b128 %0, %1, %2 offset:16"
                     :: "v"(ldsOff), "v"(aVoff), "s"(sA) : "memory");
    };
    float4 wreg[4];
    auto loadW = [&](int k0) {
        for (int it = 0; it < 4; ++it)
            wreg[it] = *reinterpret_cast<const float4*>(
                W + (size_t)(k0 + wk[it]) * ldw + bn0 + wnl[it]);
    };
    auto storeW = [&](int buf) {
        for (int it = 0; it < 4; ++it) {
            bf16_t* s = &lsB[buf][0];
            s[(wnl[it] + 0) * LSTR + wk[it]] = (bf16_t)wreg[it].x;
            s[(wnl[it] + 1) * LSTR + wk[it]] = (bf16_t)wreg[it].y;
            s[(wnl[it] + 2) * LSTR + wk[it]] = (bf16_t)wreg[it].z;
            s[(wnl[it] + 3) * LSTR + wk[it]] = (bf16_t)wreg[it].w;
        }
    };

    v8f acc[2][4];
    for (int i = 0; i < 2; ++i)
        for (int j = 0; j < 4; ++j)
            acc[i][j] = (v8f)(0.0f);

    union FragU { uint4 u[2]; v16bf v; };

    // ---- prologue: fill buffer 0 ----
    issueA(0, 0);
    loadW(0);
    storeW(0);
    asm volatile("s_wait_asynccnt 0" ::: "memory");
    __syncthreads();

    int cur = 0;
    for (int k0 = 0; k0 < Ktot; k0 += BK) {
        const int nxt = cur ^ 1;
        const bool hasNext = (k0 + BK) < Ktot;
        if (hasNext) {
            issueA(k0 + BK, nxt);   // async global->LDS, overlapped with WMMA
            loadW(k0 + BK);         // in-flight global loads
        }

        // --- fragments from current buffer ---
        FragU af[2], bfr[4];
        for (int i = 0; i < 2; ++i) {
            const bf16_t* base = &lsA[cur][(wm * 32 + i * 16 + mr) * LSTR];
            // A 16x32 bf16: lanes 0-15 K={0..7,16..23}, lanes 16-31 K={8..15,24..31}
            af[i].u[0] = *reinterpret_cast<const uint4*>(base + half * 8);
            af[i].u[1] = *reinterpret_cast<const uint4*>(base + 16 + half * 8);
        }
        for (int j = 0; j < 4; ++j) {
            const bf16_t* base = &lsB[cur][(wn * 64 + j * 16 + mr) * LSTR];
            // B 32x16 bf16: lanes 0-15 K={0..7,16..23}, lanes 16-31 K={8..15,24..31}
            bfr[j].u[0] = *reinterpret_cast<const uint4*>(base + half * 8);
            bfr[j].u[1] = *reinterpret_cast<const uint4*>(base + 16 + half * 8);
        }
        for (int i = 0; i < 2; ++i)
            for (int j = 0; j < 4; ++j)
                acc[i][j] = __builtin_amdgcn_wmma_f32_16x16x32_bf16(
                    false, af[i].v, false, bfr[j].v,
                    (short)0, acc[i][j], false, false);

        if (hasNext) storeW(nxt);
        asm volatile("s_wait_asynccnt 0" ::: "memory");
        __syncthreads();
        cur = nxt;
    }

    // ---- epilogue: C/D layout: VGPR r -> M = r + 8*half, N = mr ----
    const int rBase0 = bm0 + wm * 32 + half * 8;
    for (int i = 0; i < 2; ++i) {
        const int rb = rBase0 + i * 16;
        for (int j = 0; j < 4; ++j) {
            const int col = bn0 + wn * 64 + j * 16 + mr;
            const float bc = bias[col];
            float ew = 0.0f;
            if constexpr (EXTRA) ew = extraW[col];
            if constexpr (MODE == 1) {
                bf16_t* p = outB + (size_t)rb * ldo + col;
                for (int r = 0; r < 8; ++r) {
                    float v = acc[i][j][r] + bc;
                    if constexpr (EXTRA) v += extraX[rb + r] * ew;
                    p[(size_t)r * ldo] = (bf16_t)activate<ACT>(v);
                }
            } else {
                float* p = outF + (size_t)rb * ldo + col;
                for (int r = 0; r < 8; ++r) {
                    float v = acc[i][j][r] + bc;
                    if constexpr (EXTRA) v += extraX[rb + r] * ew;
                    v = activate<ACT>(v);
                    if constexpr (MODE == 0) p[(size_t)r * ldo] = v;
                    else p[(size_t)r * ldo] += wvec[(rb + r) * wstride + woff] * v;
                }
            }
        }
    }
}

// ---------------- small helper kernels ----------------
__global__ void k_f32_to_bf16(const float* __restrict__ x, bf16_t* __restrict__ y, int n) {
    int i = blockIdx.x * blockDim.x + threadIdx.x;
    if (i < n) y[i] = (bf16_t)x[i];
}

__global__ void k_zero(float* __restrict__ p, int n) {
    int i = blockIdx.x * blockDim.x + threadIdx.x;
    if (i < n) p[i] = 0.0f;
}

__global__ void k_i2f(const int* __restrict__ m, float* __restrict__ f, int n) {
    int i = blockIdx.x * blockDim.x + threadIdx.x;
    if (i < n) f[i] = (float)m[i];
}

// conv1d(1->4, k=3, pad=1) + flatten (channel-major) -> bf16 [B, 4*DIN]
__global__ void k_conv(const float* __restrict__ x, const float* __restrict__ ck,
                       const float* __restrict__ cb, bf16_t* __restrict__ xc) {
    int idx = blockIdx.x * blockDim.x + threadIdx.x;   // b*DIN + i
    int b = idx >> 10, i = idx & 1023;
    float xm = (i > 0)    ? x[idx - 1] : 0.0f;
    float x0 = x[idx];
    float xp = (i < 1023) ? x[idx + 1] : 0.0f;
    for (int c = 0; c < 4; ++c) {
        float v = ck[c * 3] * xm + ck[c * 3 + 1] * x0 + ck[c * 3 + 2] * xp + cb[c];
        xc[(size_t)b * (4 * DIN) + c * DIN + i] = (bf16_t)v;
    }
}

// per-row LayerNorm: f32 in -> bf16 out
__global__ __launch_bounds__(256)
void k_layernorm(const float* __restrict__ h, const float* __restrict__ g,
                 const float* __restrict__ be, bf16_t* __restrict__ out) {
    __shared__ float red[256];
    const int row = blockIdx.x, t = threadIdx.x;
    const float* hr = h + (size_t)row * DOUT;
    float s = 0.0f, sq = 0.0f;
    for (int c = t; c < DOUT; c += 256) { float v = hr[c]; s += v; sq += v * v; }
    red[t] = s; __syncthreads();
    for (int o = 128; o; o >>= 1) { if (t < o) red[t] += red[t + o]; __syncthreads(); }
    const float mean = red[0] / DOUT;
    __syncthreads();
    red[t] = sq; __syncthreads();
    for (int o = 128; o; o >>= 1) { if (t < o) red[t] += red[t + o]; __syncthreads(); }
    const float var = red[0] / DOUT - mean * mean;
    const float rs  = rsqrtf(var + 1e-5f);
    for (int c = t; c < DOUT; c += 256)
        out[(size_t)row * DOUT + c] = (bf16_t)((hr[c] - mean) * rs * g[c] + be[c]);
}

// per-channel BatchNorm (batch stats): f32 in -> bf16 out
__global__ __launch_bounds__(256)
void k_batchnorm(const float* __restrict__ h, const float* __restrict__ g,
                 const float* __restrict__ be, bf16_t* __restrict__ out) {
    __shared__ float red[256];
    const int c = blockIdx.x, t = threadIdx.x;
    float s = 0.0f, sq = 0.0f;
    for (int r = t; r < B_SZ; r += 256) { float v = h[(size_t)r * DOUT + c]; s += v; sq += v * v; }
    red[t] = s; __syncthreads();
    for (int o = 128; o; o >>= 1) { if (t < o) red[t] += red[t + o]; __syncthreads(); }
    const float mean = red[0] / B_SZ;
    __syncthreads();
    red[t] = sq; __syncthreads();
    for (int o = 128; o; o >>= 1) { if (t < o) red[t] += red[t + o]; __syncthreads(); }
    const float var = red[0] / B_SZ - mean * mean;
    const float rs  = rsqrtf(var + 1e-5f);
    for (int r = t; r < B_SZ; r += 256)
        out[(size_t)r * DOUT + c] = (bf16_t)((h[(size_t)r * DOUT + c] - mean) * rs * g[c] + be[c]);
}

// router: logits = hg[row] @ gW2 + gb2; clip(/0.7, ±10); usage softmax (atomics);
// top-4 (tie -> lowest index) + softmax over the 4 -> dense weight row w[b, 0..7]
__global__ __launch_bounds__(64)
void k_router(const float* __restrict__ hg, const float* __restrict__ W2,
              const float* __restrict__ b2, float* __restrict__ w,
              float* __restrict__ usage) {
    __shared__ float red[NEXP * 64];
    const int row = blockIdx.x, t = threadIdx.x;
    const float* hr = hg + (size_t)row * 512;
    float acc[NEXP];
    for (int e = 0; e < NEXP; ++e) acc[e] = 0.0f;
    for (int k = t; k < 512; k += 64) {
        const float hv = hr[k];
        const float* wr = W2 + k * NEXP;
        for (int e = 0; e < NEXP; ++e) acc[e] += hv * wr[e];
    }
    for (int e = 0; e < NEXP; ++e) red[e * 64 + t] = acc[e];
    __syncthreads();
    for (int o = 32; o; o >>= 1) {
        if (t < o) for (int e = 0; e < NEXP; ++e) red[e * 64 + t] += red[e * 64 + t + o];
        __syncthreads();
    }
    if (t == 0) {
        float l[NEXP];
        for (int e = 0; e < NEXP; ++e)
            l[e] = fminf(fmaxf((red[e * 64] + b2[e]) / 0.7f, -10.0f), 10.0f);
        float mx = l[0];
        for (int e = 1; e < NEXP; ++e) mx = fmaxf(mx, l[e]);
        float se = 0.0f, p[NEXP];
        for (int e = 0; e < NEXP; ++e) { p[e] = expf(l[e] - mx); se += p[e]; }
        for (int e = 0; e < NEXP; ++e) atomicAdd(&usage[e], p[e] / se);
        bool used[NEXP] = {};
        int   idxs[4];
        float vals[4];
        for (int k = 0; k < 4; ++k) {
            float best = -1e30f; int bi = 0;
            for (int e = 0; e < NEXP; ++e)
                if (!used[e] && l[e] > best) { best = l[e]; bi = e; }
            used[bi] = true; idxs[k] = bi; vals[k] = best;
        }
        float s2 = 0.0f, tw[4];
        for (int k = 0; k < 4; ++k) { tw[k] = expf(vals[k] - vals[0]); s2 += tw[k]; }
        float* wr = w + (size_t)row * NEXP;
        for (int e = 0; e < NEXP; ++e) wr[e] = 0.0f;
        for (int k = 0; k < 4; ++k) wr[idxs[k]] = tw[k] / s2;
    }
}

__global__ void k_aux(const float* __restrict__ usage, float* __restrict__ out) {
    float a = 0.0f;
    for (int e = 0; e < NEXP; ++e) {
        float u = usage[e] / (float)B_SZ;
        a += 0.125f * (logf(0.125f) - logf(u + 1e-10f));
    }
    out[0] = a * 0.125f;
}

// ---------------- launch ----------------
extern "C" void kernel_launch(void* const* d_in, const int* in_sizes, int n_in,
                              void* d_out, int out_size, void* d_ws, size_t ws_size,
                              hipStream_t stream) {
    (void)in_sizes; (void)n_in; (void)out_size; (void)ws_size;
    const float* x       = (const float*)d_in[0];
    const int*   mod     = (const int*)  d_in[1];
    const float* mlp_W1  = (const float*)d_in[2];
    const float* mlp_b1  = (const float*)d_in[3];
    const float* mlp_W2  = (const float*)d_in[4];
    const float* mlp_b2  = (const float*)d_in[5];
    const float* bn_W1   = (const float*)d_in[6];
    const float* bn_b1   = (const float*)d_in[7];
    const float* bn_g    = (const float*)d_in[8];
    const float* bn_be   = (const float*)d_in[9];
    const float* bn_W2   = (const float*)d_in[10];
    const float* bn_b2   = (const float*)d_in[11];
    const float* conv_k  = (const float*)d_in[12];
    const float* conv_b  = (const float*)d_in[13];
    const float* conv_W  = (const float*)d_in[14];
    const float* conv_bL = (const float*)d_in[15];
    const float* ln_W1   = (const float*)d_in[16];
    const float* ln_b1   = (const float*)d_in[17];
    const float* ln_g    = (const float*)d_in[18];
    const float* ln_be   = (const float*)d_in[19];
    const float* ln_W2   = (const float*)d_in[20];
    const float* ln_b2   = (const float*)d_in[21];
    const float* sh_W1   = (const float*)d_in[22];
    const float* sh_b1   = (const float*)d_in[23];
    const float* sh_g    = (const float*)d_in[24];
    const float* sh_be   = (const float*)d_in[25];
    const float* sh_W2   = (const float*)d_in[26];
    const float* sh_b2   = (const float*)d_in[27];
    const float* g_W1    = (const float*)d_in[28];
    const float* g_b1    = (const float*)d_in[29];
    const float* g_W2    = (const float*)d_in[30];
    const float* g_b2    = (const float*)d_in[31];

    char* ws = (char*)d_ws;
    bf16_t* xb    = (bf16_t*)ws;  ws += (size_t)B_SZ * DIN * 2;
    bf16_t* xc    = (bf16_t*)ws;  ws += (size_t)B_SZ * 4 * DIN * 2;
    float*  hf    = (float*)ws;   ws += (size_t)B_SZ * DOUT * 4;
    bf16_t* hb    = (bf16_t*)ws;  ws += (size_t)B_SZ * DOUT * 2;
    float*  hg    = (float*)ws;   ws += (size_t)B_SZ * 512 * 4;
    float*  w     = (float*)ws;   ws += (size_t)B_SZ * NEXP * 4;
    float*  usage = (float*)ws;   ws += 16 * 4;
    float*  modf  = (float*)ws;   ws += (size_t)B_SZ * 4;

    float* out_shared = (float*)d_out;
    float* out_priv   = out_shared + (size_t)B_SZ * DOUT;
    float* out_aux    = out_priv   + (size_t)B_SZ * DOUT;

    const dim3 gridF(DOUT / BN, B_SZ / BM);   // N=1024 GEMMs
    const dim3 gridG(512 / BN,  B_SZ / BM);   // gate N=512

    // prep
    k_f32_to_bf16<<<(B_SZ * DIN) / 256, 256, 0, stream>>>(x, xb, B_SZ * DIN);
    k_zero<<<(B_SZ * DOUT) / 256, 256, 0, stream>>>(out_priv, B_SZ * DOUT);
    k_zero<<<1, 64, 0, stream>>>(usage, NEXP);
    k_i2f<<<B_SZ / 256, 256, 0, stream>>>(mod, modf, B_SZ);

    // ---- gate: tanh(x@W1[:1024] + mod*W1[1024] + b1) ----
    gemm_bf16_wmma<2, 0, 1><<<gridG, 256, 0, stream>>>(
        xb, DIN, g_W1, 512, g_b1, hg, nullptr, 512,
        nullptr, 0, 0, g_W1 + (size_t)1024 * 512, modf, DIN);
    k_router<<<B_SZ, 64, 0, stream>>>(hg, g_W2, g_b2, w, usage);
    k_aux<<<1, 1, 0, stream>>>(usage, out_aux);

    // ---- shared expert: relu -> LN -> linear ----
    gemm_bf16_wmma<1, 0, 0><<<gridF, 256, 0, stream>>>(
        xb, DIN, sh_W1, DOUT, sh_b1, hf, nullptr, DOUT,
        nullptr, 0, 0, nullptr, nullptr, DIN);
    k_layernorm<<<B_SZ, 256, 0, stream>>>(hf, sh_g, sh_be, hb);
    gemm_bf16_wmma<0, 0, 0><<<gridF, 256, 0, stream>>>(
        hb, DOUT, sh_W2, DOUT, sh_b2, out_shared, nullptr, DOUT,
        nullptr, 0, 0, nullptr, nullptr, DOUT);

    // ---- experts 0-1: Linear-ReLU-Linear ----
    for (int e = 0; e < 2; ++e) {
        gemm_bf16_wmma<1, 1, 0><<<gridF, 256, 0, stream>>>(
            xb, DIN, mlp_W1 + (size_t)e * DIN * DOUT, DOUT,
            mlp_b1 + (size_t)e * DOUT, nullptr, hb, DOUT,
            nullptr, 0, 0, nullptr, nullptr, DIN);
        gemm_bf16_wmma<0, 2, 0><<<gridF, 256, 0, stream>>>(
            hb, DOUT, mlp_W2 + (size_t)e * DOUT * DOUT, DOUT,
            mlp_b2 + (size_t)e * DOUT, out_priv, nullptr, DOUT,
            w, NEXP, 0 + e, nullptr, nullptr, DOUT);
    }
    // ---- experts 2-3: Linear-Tanh-BN-Linear ----
    for (int e = 0; e < 2; ++e) {
        gemm_bf16_wmma<2, 0, 0><<<gridF, 256, 0, stream>>>(
            xb, DIN, bn_W1 + (size_t)e * DIN * DOUT, DOUT,
            bn_b1 + (size_t)e * DOUT, hf, nullptr, DOUT,
            nullptr, 0, 0, nullptr, nullptr, DIN);
        k_batchnorm<<<DOUT, 256, 0, stream>>>(
            hf, bn_g + (size_t)e * DOUT, bn_be + (size_t)e * DOUT, hb);
        gemm_bf16_wmma<0, 2, 0><<<gridF, 256, 0, stream>>>(
            hb, DOUT, bn_W2 + (size_t)e * DOUT * DOUT, DOUT,
            bn_b2 + (size_t)e * DOUT, out_priv, nullptr, DOUT,
            w, NEXP, 2 + e, nullptr, nullptr, DOUT);
    }
    // ---- experts 4-5: Conv1d-Flatten-Linear-GELU ----
    for (int e = 0; e < 2; ++e) {
        k_conv<<<(B_SZ * DIN) / 256, 256, 0, stream>>>(
            x, conv_k + (size_t)e * 12, conv_b + (size_t)e * 4, xc);
        gemm_bf16_wmma<4, 2, 0><<<gridF, 256, 0, stream>>>(
            xc, 4 * DIN, conv_W + (size_t)e * 4 * DIN * DOUT, DOUT,
            conv_bL + (size_t)e * DOUT, out_priv, nullptr, DOUT,
            w, NEXP, 4 + e, nullptr, nullptr, 4 * DIN);
    }
    // ---- experts 6-7: Linear-SiLU-LN-Linear ----
    for (int e = 0; e < 2; ++e) {
        gemm_bf16_wmma<3, 0, 0><<<gridF, 256, 0, stream>>>(
            xb, DIN, ln_W1 + (size_t)e * DIN * DOUT, DOUT,
            ln_b1 + (size_t)e * DOUT, hf, nullptr, DOUT,
            nullptr, 0, 0, nullptr, nullptr, DIN);
        k_layernorm<<<B_SZ, 256, 0, stream>>>(
            hf, ln_g + (size_t)e * DOUT, ln_be + (size_t)e * DOUT, hb);
        gemm_bf16_wmma<0, 2, 0><<<gridF, 256, 0, stream>>>(
            hb, DOUT, ln_W2 + (size_t)e * DOUT * DOUT, DOUT,
            ln_b2 + (size_t)e * DOUT, out_priv, nullptr, DOUT,
            w, NEXP, 6 + e, nullptr, nullptr, DOUT);
    }
}
